// SpatialAggregator_79559974191158
// MI455X (gfx1250) — compile-verified, matrix operand order checked by reading, not run on hardware
//
#include <hip/hip_runtime.h>
#include <math.h>

// ---------------------------------------------------------------------------
// Types for CDNA5 WMMA (wave32): v_wmma_f32_16x16x32_bf16
// ---------------------------------------------------------------------------
typedef __attribute__((ext_vector_type(16))) __bf16 v16bf;
typedef __attribute__((ext_vector_type(8)))  float  v8f;

#define TOK    49     // tokens per 7x7 window
#define NPAD   64     // padded tokens (4 x 16)
#define NH     8      // heads
#define DH     16     // head dim
#define DMODEL 128
#define DCAT   384    // DMODEL + GDIM
#define GD     256
#define HW     56
#define IMG_PIX (HW*HW)           // 3136
#define N_IMG  64                 // B*C = 2*32
#define N_TOKENS ((long long)N_IMG * IMG_PIX)   // 200704
#define X_ELEMS  (N_TOKENS * DMODEL)            // 25690112

// hardware f32 -> bf16 (RNE) conversion
__device__ __forceinline__ __bf16 f2bf(float x) { return (__bf16)x; }

__device__ __forceinline__ v8f zero8() {
  v8f z;
#pragma unroll
  for (int i = 0; i < 8; ++i) z[i] = 0.0f;
  return z;
}

// A-fragment (and B-fragment when B^T is stored row-major N x K):
// lane l: m = l&15, k-half = (l>>4)*8; elems 0..7 -> k=kh+j, elems 8..15 -> k=16+kh+j
__device__ __forceinline__ v16bf load_frag_row(const __bf16* src, int ld) {
  int lane = threadIdx.x & 31;
  int m = lane & 15;
  int kh = (lane >> 4) * 8;
  const __bf16* row = src + m * ld;
  v16bf a;
#pragma unroll
  for (int j = 0; j < 8; ++j) a[j] = row[kh + j];
#pragma unroll
  for (int j = 0; j < 8; ++j) a[8 + j] = row[16 + kh + j];
  return a;
}

// Same but only k<16 is valid (head-dim 16), upper half zero-padded.
__device__ __forceinline__ v16bf load_frag_k16(const __bf16* src, int ld) {
  int lane = threadIdx.x & 31;
  int m = lane & 15;
  int kh = (lane >> 4) * 8;
  const __bf16* row = src + m * ld;
  v16bf a;
#pragma unroll
  for (int j = 0; j < 8; ++j) a[j] = row[kh + j];          // k in [0,16)
#pragma unroll
  for (int j = 0; j < 8; ++j) a[8 + j] = f2bf(0.0f);       // k in [16,32)
  return a;
}

// B-fragment when B^T(n,k) = src[k*ld + n] (i.e. B stored K-major, like V slice)
__device__ __forceinline__ v16bf load_frag_col(const __bf16* src, int ld) {
  int lane = threadIdx.x & 31;
  int n = lane & 15;
  int kh = (lane >> 4) * 8;
  v16bf a;
#pragma unroll
  for (int j = 0; j < 8; ++j) a[j] = src[(kh + j) * ld + n];
#pragma unroll
  for (int j = 0; j < 8; ++j) a[8 + j] = src[(16 + kh + j) * ld + n];
  return a;
}

#define WMMA_BF16(A, B, C) \
  __builtin_amdgcn_wmma_f32_16x16x32_bf16(false, (A), false, (B), (short)0, (C), false, false)

__device__ __forceinline__ int regid(int c) {  // shift-mask region along one axis
  return c < (HW - 7) ? 0 : (c < (HW - 3) ? 1 : 2);
}

// ---------------------------------------------------------------------------
// Weight transpose + f32 -> bf16:  src (K x N) row-major  ->  dst (N x K)
// ---------------------------------------------------------------------------
__global__ void k_transpose_bf16(const float* __restrict__ src, __bf16* __restrict__ dst,
                                 int K, int N) {
  int id = blockIdx.x * 256 + threadIdx.x;
  if (id < K * N) {
    int k = id / N, n = id - k * N;
    dst[n * K + k] = f2bf(src[id]);
  }
}

// x (B=2, D=128, C=32, H, W) -> xA token-major (((b*32+c)*56+h)*56+w, 128)
__global__ void k_init_x(const float* __restrict__ x, float* __restrict__ xA) {
  long long id = (long long)blockIdx.x * 256 + threadIdx.x;
  if (id < X_ELEMS) {
    int d = (int)(id & 127);
    long long t = id >> 7;
    int w = (int)(t % HW); t /= HW;
    int h = (int)(t % HW); t /= HW;
    int c = (int)(t % 32);
    int b = (int)(t / 32);
    size_t s = ((((size_t)b * 128 + d) * 32 + c) * HW + h) * HW + w;
    xA[id] = x[s];
  }
}

// xA token-major -> out (B, D, C, H, W)
__global__ void k_final(const float* __restrict__ xA, float* __restrict__ out) {
  long long id = (long long)blockIdx.x * 256 + threadIdx.x;
  if (id < X_ELEMS) {
    long long t = id;
    int w = (int)(t % HW); t /= HW;
    int h = (int)(t % HW); t /= HW;
    int c = (int)(t % 32); t /= 32;
    int d = (int)(t % 128);
    int b = (int)(t / 128);
    size_t s = ((((size_t)b * 32 + c) * HW + h) * HW + w) * DMODEL + d;
    out[id] = xA[s];
  }
}

// Uniform-control QKV projection sub-GEMM: 32 output tiles across 8 waves.
template <int KD>
__device__ __forceinline__ void qkv_gemm(const __bf16* sWin, const __bf16* __restrict__ wT,
                                         const float* __restrict__ bias, float sc,
                                         __bf16* dst, int wvid, int lane) {
  const int lh = lane >> 4, ln = lane & 15;
#pragma unroll
  for (int tt = 0; tt < 4; ++tt) {
    int t = wvid + tt * 8;           // uniform (wvid is SGPR)
    int mt = t >> 3, nt = t & 7;
    v8f acc = zero8();
#pragma unroll
    for (int k = 0; k < (KD >> 5); ++k) {
      v16bf a = load_frag_row(&sWin[(mt * 16) * DCAT + k * 32], DCAT);
      v16bf b = load_frag_row(&wT[(nt * 16) * KD + k * 32], KD);
      acc = WMMA_BF16(a, b, acc);
    }
    int n = nt * 16 + ln;
    float bn = bias[n];
    int m0 = mt * 16 + 8 * lh;
#pragma unroll
    for (int r = 0; r < 8; ++r) dst[(m0 + r) * DMODEL + n] = f2bf((acc[r] + bn) * sc);
  }
}

// ---------------------------------------------------------------------------
// Fused windowed attention block: one workgroup per 7x7 window (padded to 64)
// ---------------------------------------------------------------------------
__global__ __launch_bounds__(256) void k_attn(
    const float* __restrict__ xin, const float* __restrict__ guid,
    const float* __restrict__ gng, const float* __restrict__ gnb,
    const float* __restrict__ n1g, const float* __restrict__ n1b,
    const __bf16* __restrict__ wqT, const float* __restrict__ bq,
    const __bf16* __restrict__ wkT, const float* __restrict__ bk,
    const __bf16* __restrict__ wvT, const float* __restrict__ bv,
    const __bf16* __restrict__ wpT, const float* __restrict__ bpj,
    const float* __restrict__ rpb,
    float* __restrict__ xout, int shift) {
  // smem0: stage1/2 -> window concat (64x384 bf16 = 48KB); stage3 -> per-wave P (8x8KB)
  __shared__ __align__(16) char smem0[65536];
  __shared__ __bf16 sQ[NPAD * DMODEL];
  __shared__ __bf16 sK[NPAD * DMODEL];
  __shared__ __bf16 sV[NPAD * DMODEL];
  __shared__ __bf16 sOb[NPAD * DMODEL];
  __shared__ float  sRpb[169 * NH];           // rel-pos bias table

  __bf16* sWin = reinterpret_cast<__bf16*>(smem0);

  const int img  = blockIdx.x >> 6;
  const int widx = blockIdx.x & 63;
  const int wy = widx >> 3, wx = widx & 7;
  const int lane = threadIdx.x & 31;
  const int wvid = __builtin_amdgcn_readfirstlane(threadIdx.x >> 5);  // uniform -> SGPR
  const int lh = lane >> 4, ln = lane & 15;

  // stage rel-pos bias into LDS
  for (int i = threadIdx.x; i < 169 * NH; i += 256) sRpb[i] = rpb[i];

  // ---- Stage 1: build concat window [LN(x) | LN(guidance)] with roll ----
  for (int r = wvid; r < NPAD; r += 8) {
    if (r < TOK) {
      int ty = r / 7, tx = r % 7;
      int hs = wy * 7 + ty + shift; if (hs >= HW) hs -= HW;
      int ws_ = wx * 7 + tx + shift; if (ws_ >= HW) ws_ -= HW;
      size_t pix = (size_t)img * IMG_PIX + hs * HW + ws_;

      const float* xrow = xin + pix * DMODEL;
      float v0[4], s = 0.f, s2 = 0.f;
#pragma unroll
      for (int j = 0; j < 4; ++j) { float t = xrow[lane + 32 * j]; v0[j] = t; s += t; s2 += t * t; }
#pragma unroll
      for (int m = 16; m >= 1; m >>= 1) { s += __shfl_xor(s, m, 32); s2 += __shfl_xor(s2, m, 32); }
      float mu = s * (1.f / 128.f);
      float rs = rsqrtf(s2 * (1.f / 128.f) - mu * mu + 1e-5f);
#pragma unroll
      for (int j = 0; j < 4; ++j) {
        int d = lane + 32 * j;
        sWin[r * DCAT + d] = f2bf((v0[j] - mu) * rs * n1g[d] + n1b[d]);
      }

      const float* grow = guid + pix * GD;
      float g0[8]; s = 0.f; s2 = 0.f;
#pragma unroll
      for (int j = 0; j < 8; ++j) { float t = grow[lane + 32 * j]; g0[j] = t; s += t; s2 += t * t; }
#pragma unroll
      for (int m = 16; m >= 1; m >>= 1) { s += __shfl_xor(s, m, 32); s2 += __shfl_xor(s2, m, 32); }
      mu = s * (1.f / 256.f);
      rs = rsqrtf(s2 * (1.f / 256.f) - mu * mu + 1e-5f);
#pragma unroll
      for (int j = 0; j < 8; ++j) {
        int d = lane + 32 * j;
        sWin[r * DCAT + DMODEL + d] = f2bf((g0[j] - mu) * rs * gng[d] + gnb[d]);
      }
    } else {
      for (int d = lane; d < DCAT; d += 32) sWin[r * DCAT + d] = f2bf(0.0f);
    }
  }
  __syncthreads();

  // ---- Stage 2: Q/K/V projections (uniform control, compile-time K depth) ----
  qkv_gemm<DCAT>(sWin, wqT, bq, 0.25f, sQ, wvid, lane);   // scale = hd^-0.5
  qkv_gemm<DCAT>(sWin, wkT, bk, 1.0f, sK, wvid, lane);
  qkv_gemm<DMODEL>(sWin, wvT, bv, 1.0f, sV, wvid, lane);  // V reads first 128 cols only
  __syncthreads();

  // ---- Stage 3: per-head attention (wave wvid = head wvid) ----
  {
    const int h = wvid;
    __bf16* sPb = reinterpret_cast<__bf16*>(smem0) + h * (NPAD * NPAD); // aliases dead sWin
    v8f S[4][4];
#pragma unroll
    for (int mt = 0; mt < 4; ++mt) {
      v16bf a = load_frag_k16(&sQ[(mt * 16) * DMODEL + h * DH], DMODEL);
#pragma unroll
      for (int nt = 0; nt < 4; ++nt) {
        v16bf b = load_frag_k16(&sK[(nt * 16) * DMODEL + h * DH], DMODEL);
        v8f z = zero8();
        S[mt][nt] = WMMA_BF16(a, b, z);
      }
    }
    // relative position bias + shift mask
#pragma unroll
    for (int mt = 0; mt < 4; ++mt) {
#pragma unroll
      for (int nt = 0; nt < 4; ++nt) {
        int j = nt * 16 + ln;
        int yj = j / 7, xj = j % 7;
#pragma unroll
        for (int r = 0; r < 8; ++r) {
          int i = mt * 16 + 8 * lh + r;
          float v = S[mt][nt][r];
          if (i < TOK && j < TOK) {
            int yi = i / 7, xi = i % 7;
            int ridx = (yi - yj + 6) * 13 + (xi - xj + 6);
            v += sRpb[ridx * NH + h];
            if (shift > 0) {
              int ci = regid(wy * 7 + yi) * 3 + regid(wx * 7 + xi);
              int cj = regid(wy * 7 + yj) * 3 + regid(wx * 7 + xj);
              if (ci != cj) v -= 100.0f;
            }
          } else {
            v = -1e30f;   // padded columns -> exp 0
          }
          S[mt][nt][r] = v;
        }
      }
    }
    // row softmax: columns live in ln (bits 0..3) across 4 nt tiles
#pragma unroll
    for (int mt = 0; mt < 4; ++mt) {
#pragma unroll
      for (int r = 0; r < 8; ++r) {
        float m_ = fmaxf(fmaxf(S[mt][0][r], S[mt][1][r]), fmaxf(S[mt][2][r], S[mt][3][r]));
#pragma unroll
        for (int msk = 8; msk >= 1; msk >>= 1) m_ = fmaxf(m_, __shfl_xor(m_, msk, 32));
        float e0 = __expf(S[mt][0][r] - m_);
        float e1 = __expf(S[mt][1][r] - m_);
        float e2 = __expf(S[mt][2][r] - m_);
        float e3 = __expf(S[mt][3][r] - m_);
        float sum = e0 + e1 + e2 + e3;
#pragma unroll
        for (int msk = 8; msk >= 1; msk >>= 1) sum += __shfl_xor(sum, msk, 32);
        float inv = 1.0f / (sum + 1e-30f);
        int row = mt * 16 + 8 * lh + r;
        sPb[row * NPAD + 0 * 16 + ln] = f2bf(e0 * inv);
        sPb[row * NPAD + 1 * 16 + ln] = f2bf(e1 * inv);
        sPb[row * NPAD + 2 * 16 + ln] = f2bf(e2 * inv);
        sPb[row * NPAD + 3 * 16 + ln] = f2bf(e3 * inv);
      }
    }
    // out_h = P @ V_h   (K = 64 -> 2 wmma steps)
#pragma unroll
    for (int mt = 0; mt < 4; ++mt) {
      v8f o = zero8();
#pragma unroll
      for (int k = 0; k < 2; ++k) {
        v16bf a = load_frag_row(&sPb[(mt * 16) * NPAD + k * 32], NPAD);
        v16bf b = load_frag_col(&sV[(k * 32) * DMODEL + h * DH], DMODEL);
        o = WMMA_BF16(a, b, o);
      }
      int m0 = mt * 16 + 8 * lh;
#pragma unroll
      for (int r = 0; r < 8; ++r)
        sOb[(m0 + r) * DMODEL + h * DH + ln] = f2bf(o[r]);
    }
  }
  __syncthreads();

  // ---- Stage 4: output projection + residual, inverse roll = same coords ----
  {
    const int nt = wvid;
#pragma unroll
    for (int mt = 0; mt < 4; ++mt) {
      v8f acc = zero8();
#pragma unroll
      for (int k = 0; k < 4; ++k) {
        v16bf a = load_frag_row(&sOb[(mt * 16) * DMODEL + k * 32], DMODEL);
        v16bf b = load_frag_row(&wpT[(nt * 16) * DMODEL + k * 32], DMODEL);
        acc = WMMA_BF16(a, b, acc);
      }
      int n = nt * 16 + ln;
      float bn = bpj[n];
#pragma unroll
      for (int r = 0; r < 8; ++r) {
        int i = mt * 16 + 8 * lh + r;
        if (i < TOK) {
          int ty = i / 7, tx = i % 7;
          int hs = wy * 7 + ty + shift; if (hs >= HW) hs -= HW;
          int ws_ = wx * 7 + tx + shift; if (ws_ >= HW) ws_ -= HW;
          size_t t = (size_t)img * IMG_PIX + hs * HW + ws_;
          xout[t * DMODEL + n] = xin[t * DMODEL + n] + acc[r] + bn;
        }
      }
    }
  }
}

// ---------------------------------------------------------------------------
// Fused MLP block (in-place): x += GELU(LN(x) @ w1 + b1) @ w2
// 64 tokens per workgroup; hidden 512 processed in 4 tiles of 128.
// ---------------------------------------------------------------------------
__global__ __launch_bounds__(256) void k_mlp(
    float* __restrict__ x,
    const float* __restrict__ n2g, const float* __restrict__ n2b,
    const __bf16* __restrict__ w1T, const float* __restrict__ b1,
    const __bf16* __restrict__ w2T) {
  __shared__ __bf16 sXn[64 * DMODEL];
  __shared__ __bf16 sH[64 * DMODEL];
  const int lane = threadIdx.x & 31;
  const int wvid = __builtin_amdgcn_readfirstlane(threadIdx.x >> 5);  // uniform -> SGPR
  const int lh = lane >> 4, ln = lane & 15;
  const size_t base = (size_t)blockIdx.x * 64;

  for (int r = wvid; r < 64; r += 8) {
    const float* xr = x + (base + r) * DMODEL;
    float v0[4], s = 0.f, s2 = 0.f;
#pragma unroll
    for (int j = 0; j < 4; ++j) { float t = xr[lane + 32 * j]; v0[j] = t; s += t; s2 += t * t; }
#pragma unroll
    for (int m = 16; m >= 1; m >>= 1) { s += __shfl_xor(s, m, 32); s2 += __shfl_xor(s2, m, 32); }
    float mu = s * (1.f / 128.f);
    float rs = rsqrtf(s2 * (1.f / 128.f) - mu * mu + 1e-5f);
#pragma unroll
    for (int j = 0; j < 4; ++j) {
      int d = lane + 32 * j;
      sXn[r * DMODEL + d] = f2bf((v0[j] - mu) * rs * n2g[d] + n2b[d]);
    }
  }
  __syncthreads();

  v8f acc[4];
#pragma unroll
  for (int mt = 0; mt < 4; ++mt) acc[mt] = zero8();

#pragma unroll
  for (int ht = 0; ht < 4; ++ht) {
    // H tile: wave wvid owns hidden sub-column wvid (16 wide) for all mt
#pragma unroll
    for (int mt = 0; mt < 4; ++mt) {
      v8f hacc = zero8();
#pragma unroll
      for (int k = 0; k < 4; ++k) {
        v16bf a = load_frag_row(&sXn[(mt * 16) * DMODEL + k * 32], DMODEL);
        v16bf b = load_frag_row(&w1T[(size_t)(ht * 128 + wvid * 16) * DMODEL + k * 32], DMODEL);
        hacc = WMMA_BF16(a, b, hacc);
      }
      int hcol = ht * 128 + wvid * 16 + ln;
      float bn = b1[hcol];
      int m0 = mt * 16 + 8 * lh;
#pragma unroll
      for (int r = 0; r < 8; ++r) {
        float t = hacc[r] + bn;
        t = 0.5f * t * (1.0f + erff(t * 0.70710678118654752f));   // exact GELU
        sH[(m0 + r) * DMODEL + wvid * 16 + ln] = f2bf(t);
      }
    }
    __syncthreads();
    // acc += H_tile @ w2 slice  (wave wvid owns output sub-column wvid)
#pragma unroll
    for (int mt = 0; mt < 4; ++mt) {
#pragma unroll
      for (int k = 0; k < 4; ++k) {
        v16bf a = load_frag_row(&sH[(mt * 16) * DMODEL + k * 32], DMODEL);
        v16bf b = load_frag_row(&w2T[(size_t)(wvid * 16) * 512 + ht * 128 + k * 32], 512);
        acc[mt] = WMMA_BF16(a, b, acc[mt]);
      }
    }
    __syncthreads();
  }

#pragma unroll
  for (int mt = 0; mt < 4; ++mt) {
    int n = wvid * 16 + ln;
    int m0 = mt * 16 + 8 * lh;
#pragma unroll
    for (int r = 0; r < 8; ++r) {
      size_t i = base + m0 + r;
      x[i * DMODEL + n] = x[i * DMODEL + n] + acc[mt][r];
    }
  }
}

// ---------------------------------------------------------------------------
// Host launcher
// ---------------------------------------------------------------------------
extern "C" void kernel_launch(void* const* d_in, const int* in_sizes, int n_in,
                              void* d_out, int out_size, void* d_ws, size_t ws_size,
                              hipStream_t stream) {
  (void)in_sizes; (void)n_in; (void)out_size; (void)ws_size;
  const float* x        = (const float*)d_in[0];
  const float* guidance = (const float*)d_in[1];
  const float* gnorm_g  = (const float*)d_in[2];
  const float* gnorm_b  = (const float*)d_in[3];
  // per-block params start at 4; 17 entries per block:
  // 0 n1g 1 n1b 2 wq 3 bq 4 wk 5 bk 6 wv 7 bv 8 wp 9 bp 10 rpb 11 n2g 12 n2b 13 w1 14 b1 15 w2 16 b2
  auto P = [&](int blk, int i) { return (const float*)d_in[4 + blk * 17 + i]; };

  char* ws = (char*)d_ws;
  const size_t XBYTES = (size_t)X_ELEMS * sizeof(float);   // ~102.8 MB each
  float* xA = (float*)ws;
  float* xB = (float*)(ws + XBYTES);
  __bf16* wflat = (__bf16*)(ws + 2 * XBYTES);
  const size_t WBLK = 49152 + 49152 + 16384 + 16384 + 65536 + 65536;  // elems per block

  // repack x to token-major
  {
    int blocks = (int)((X_ELEMS + 255) / 256);
    k_init_x<<<blocks, 256, 0, stream>>>(x, xA);
  }
  // transpose+convert weights for both blocks
  for (int blk = 0; blk < 2; ++blk) {
    __bf16* wq = wflat + (size_t)blk * WBLK;
    __bf16* wk = wq + 49152;
    __bf16* wv = wk + 49152;
    __bf16* wp = wv + 16384;
    __bf16* w1 = wp + 16384;
    __bf16* w2 = w1 + 65536;
    k_transpose_bf16<<<(49152 + 255) / 256, 256, 0, stream>>>(P(blk, 2),  wq, 384, 128);
    k_transpose_bf16<<<(49152 + 255) / 256, 256, 0, stream>>>(P(blk, 4),  wk, 384, 128);
    k_transpose_bf16<<<(16384 + 255) / 256, 256, 0, stream>>>(P(blk, 6),  wv, 128, 128);
    k_transpose_bf16<<<(16384 + 255) / 256, 256, 0, stream>>>(P(blk, 8),  wp, 128, 128);
    k_transpose_bf16<<<(65536 + 255) / 256, 256, 0, stream>>>(P(blk, 13), w1, 128, 512);
    k_transpose_bf16<<<(65536 + 255) / 256, 256, 0, stream>>>(P(blk, 15), w2, 512, 128);
  }

  const int nWin = N_IMG * 64;            // 4096 windows
  const int nMlp = (int)(N_TOKENS / 64);  // 3136

  for (int blk = 0; blk < 2; ++blk) {
    __bf16* wq = wflat + (size_t)blk * WBLK;
    __bf16* wk = wq + 49152;
    __bf16* wv = wk + 49152;
    __bf16* wp = wv + 16384;
    __bf16* w1 = wp + 16384;
    __bf16* w2 = w1 + 65536;
    float* xi = (blk == 0) ? xA : xB;
    float* xo = (blk == 0) ? xB : xA;
    int shift = (blk == 0) ? 0 : 3;
    k_attn<<<nWin, 256, 0, stream>>>(
        xi, guidance, gnorm_g, gnorm_b,
        P(blk, 0), P(blk, 1),
        wq, P(blk, 3), wk, P(blk, 5), wv, P(blk, 7), wp, P(blk, 9),
        P(blk, 10), xo, shift);
    k_mlp<<<nMlp, 256, 0, stream>>>(xo, P(blk, 11), P(blk, 12), w1, P(blk, 14), w2);
  }

  {
    int blocks = (int)((X_ELEMS + 255) / 256);
    k_final<<<blocks, 256, 0, stream>>>(xA, (float*)d_out);
  }
}
